// SNNPolicyActor_51780125721087
// MI455X (gfx1250) — matrix operand order repaired and checked by reading, not compiled, not run on hardware
//
#include <hip/hip_runtime.h>

// ---------------------------------------------------------------------------
// MI455X (gfx1250) fused SNN policy actor.
// bf16 WMMA (v_wmma_f32_16x16x32_bf16) for all three GEMMs, LIF scan fused
// into GEMM epilogues in registers. wave32, 256-thread blocks (8 waves).
// All f32->bf16 conversion hoisted into one-shot convert kernels so the GEMM
// main loops are pure {global_load_b128 -> ds_store -> ds_load -> v_wmma}.
// Workspace layout (bytes):
//   [0,        8388608)   W2  bf16 [2048,2048]
//   [8388608, 25165824)   S1 (spikes1) bf16 [4096,2048]
//   [25165824,41943040)   S2 (act2)    bf16 [4096,2048]
//   [41943040,44040192)   obs bf16 [4096,256]
//   [44040192,45088768)   W1  bf16 [2048,256]
//   [45088768,45219840)   W3  bf16 [32,2048]
// ---------------------------------------------------------------------------

typedef __attribute__((ext_vector_type(16))) __bf16 v16bf;
typedef __attribute__((ext_vector_type(8)))  __bf16 v8bf;
typedef __attribute__((ext_vector_type(2)))  __bf16 v2bf;
typedef __attribute__((ext_vector_type(8)))  float  v8f;
typedef __attribute__((ext_vector_type(2)))  float  v2f;

#define LDSROW 40   // 32 K-elements + 8 pad (bank stagger); 80B per row

// --- f32 -> bf16 (RNE) via fptrunc: lets the backend pick v_cvt_pk_bf16_f32
// --- if gfx1250 has it; falls back to the standard expansion otherwise.
__device__ __forceinline__ unsigned short f2bf(float f) {
  union { __bf16 b; unsigned short u; } c;
  c.b = (__bf16)f;
  return c.u;
}

__device__ __forceinline__ unsigned int f2bf_pk(float lo, float hi) {
  v2f x = {lo, hi};
  union { v2bf v; unsigned int u; } c;
  c.v = __builtin_convertvector(x, v2bf);
  return c.u;
}

__device__ __forceinline__ v8f wmma_bf16(v16bf a, v16bf b, v8f c) {
  return __builtin_amdgcn_wmma_f32_16x16x32_bf16(
      /*neg_a=*/false, a, /*neg_b=*/false, b,
      /*c_mod=*/(short)0, c, /*reuse_a=*/false, /*reuse_b=*/false);
}

// A-fragment (16x32 bf16, ISA 7.12.2): lane<16 -> M=lane, K={k0..+7, k0+16..+23};
// lane>=16 -> same M, K shifted by 8.
__device__ __forceinline__ v16bf load_fragA(const unsigned short* tile, int row0) {
  const int lane = threadIdx.x & 31;
  const unsigned short* p = tile + (row0 + (lane & 15)) * LDSROW + ((lane >> 4) * 8);
  union { v16bf v; v8bf h[2]; } u;
  u.h[0] = *(const v8bf*)(p);
  u.h[1] = *(const v8bf*)(p + 16);
  return u.v;
}

// B-fragment (32x16 bf16): lane<16 -> N=lane, K={k0..+7, k0+16..+23}; lane>=16 -> K shifted by 8.
__device__ __forceinline__ v16bf load_fragB(const unsigned short* tile, int row0) {
  const int lane = threadIdx.x & 31;
  const unsigned short* p = tile + (row0 + (lane & 15)) * LDSROW + ((lane >> 4) * 8);
  union { v16bf v; v8bf h[2]; } u;
  u.h[0] = *(const v8bf*)(p);
  u.h[1] = *(const v8bf*)(p + 16);
  return u.v;
}

// B-fragment straight from global bf16 (row-major [N,K], ld elements per row).
__device__ __forceinline__ v16bf load_fragB_g(const unsigned short* __restrict__ G,
                                              long ld, int row0, int k0) {
  const int lane = threadIdx.x & 31;
  const unsigned short* p = G + (long)(row0 + (lane & 15)) * ld + k0 + ((lane >> 4) * 8);
  union { v16bf v; v8bf h[2]; } u;
  u.h[0] = *(const v8bf*)(p);
  u.h[1] = *(const v8bf*)(p + 16);
  return u.v;
}

// --- Global->register tile load (128 rows x 32 K per block, 256 threads) ---
__device__ __forceinline__ void gload_bf_tile(const unsigned short* __restrict__ G, long ld,
                                              int row0, int k0, v8bf& x0, v8bf& x1) {
  const int t = threadIdx.x;
  const int r = t >> 1, c = (t & 1) * 16;
  const unsigned short* p = G + (long)(row0 + r) * ld + k0 + c;
  x0 = *(const v8bf*)(p);
  x1 = *(const v8bf*)(p + 8);
}

// --- Register->LDS tile store ---
__device__ __forceinline__ void stor_bf_tile(unsigned short* tile, const v8bf& x0, const v8bf& x1) {
  const int t = threadIdx.x;
  const int r = t >> 1, c = (t & 1) * 16;
  *(v8bf*)(tile + r * LDSROW + c)     = x0;
  *(v8bf*)(tile + r * LDSROW + c + 8) = x1;
}

// --- LIF scan over 8 register elements (DECAY=0.5, THRESHOLD=1.0, hard reset)
__device__ __forceinline__ void lif_scan8(const float* cur, int T, float* rate, float* mlast) {
  float m[8], s[8];
#pragma unroll
  for (int e = 0; e < 8; ++e) { m[e] = 0.0f; s[e] = 0.0f; }
  for (int t = 0; t < T; ++t) {
#pragma unroll
    for (int e = 0; e < 8; ++e) {
      m[e] = 0.5f * m[e] + cur[e];
      float sp = (m[e] >= 1.0f) ? 1.0f : 0.0f;
      s[e] += sp;
      m[e] -= m[e] * sp;                 // hard reset, branchless
    }
  }
  const float invT = 1.0f / (float)T;
#pragma unroll
  for (int e = 0; e < 8; ++e) { rate[e] = s[e] * invT; mlast[e] = m[e]; }
}

// ---------------------------------------------------------------------------
// Kernel 0: f32 -> bf16 convert, 2 elements per thread (packed cvt path)
// ---------------------------------------------------------------------------
__global__ void __launch_bounds__(256) cvt_f32_bf16(const float* __restrict__ in,
                                                    unsigned int* __restrict__ out, int npairs) {
  int i = blockIdx.x * 256 + threadIdx.x;
  if (i < npairs) out[i] = f2bf_pk(in[2 * i], in[2 * i + 1]);
}

// ---------------------------------------------------------------------------
// Kernels 1 & 2: C = A @ Bw^T + bias ; LIF ; store bf16 activation.
//   MODE 1: store spike rate               (layer 1)
//   MODE 2: store 0.5*(rate + membrane)    (layer 2)
// Block tile 128x128, wave tile 64x32 (4x2 accumulators), double-buffered LDS,
// next-tile global loads issued under the WMMA block, prefetch 2 tiles ahead.
// grid = (Brows/128, 2048/128), 256 threads.
// ---------------------------------------------------------------------------
template <int KSTEPS, int MODE>
__global__ void __launch_bounds__(256) snn_gemm_lif(
    const unsigned short* __restrict__ A,  long lda,
    const unsigned short* __restrict__ Bw, long ldb,
    const float* __restrict__ bias_v, const int* __restrict__ twp,
    unsigned short* __restrict__ Out) {
  __shared__ __align__(16) unsigned short At[2][128 * LDSROW];
  __shared__ __align__(16) unsigned short Bt[2][128 * LDSROW];
  const int B0 = blockIdx.x * 128, H0 = blockIdx.y * 128;
  const int wave = threadIdx.x >> 5, lane = threadIdx.x & 31;
  const int m0 = (wave >> 2) * 64, n0 = (wave & 3) * 32;  // wave tile: 64x32

  const v8f vzero = {0.f, 0.f, 0.f, 0.f, 0.f, 0.f, 0.f, 0.f};
  v8f acc[4][2];
#pragma unroll
  for (int i = 0; i < 4; ++i)
#pragma unroll
    for (int j = 0; j < 2; ++j) acc[i][j] = vzero;

  v8bf a0, a1, bq0, bq1;
  gload_bf_tile(A,  lda, B0, 0, a0, a1);
  gload_bf_tile(Bw, ldb, H0, 0, bq0, bq1);
  int p = 0;
  for (int kt = 0; kt < KSTEPS; ++kt) {
    stor_bf_tile(At[p], a0, a1);
    stor_bf_tile(Bt[p], bq0, bq1);
    __syncthreads();
    if (kt + 1 < KSTEPS) {                     // loads in flight during compute
      gload_bf_tile(A,  lda, B0, (kt + 1) * 32, a0, a1);
      gload_bf_tile(Bw, ldb, H0, (kt + 1) * 32, bq0, bq1);
    }
    if (kt + 2 < KSTEPS) {                     // global_prefetch_b8, 2 tiles ahead
      const int r = threadIdx.x >> 1;
      __builtin_prefetch(A  + (long)(B0 + r) * lda + (kt + 2) * 32, 0, 0);
      __builtin_prefetch(Bw + (long)(H0 + r) * ldb + (kt + 2) * 32, 0, 0);
    }
    v16bf af[4], bfr[2];
#pragma unroll
    for (int i = 0; i < 4; ++i) af[i] = load_fragA(At[p], m0 + i * 16);
#pragma unroll
    for (int j = 0; j < 2; ++j) bfr[j] = load_fragB(Bt[p], n0 + j * 16);
#pragma unroll
    for (int i = 0; i < 4; ++i)
#pragma unroll
      for (int j = 0; j < 2; ++j) acc[i][j] = wmma_bf16(af[i], bfr[j], acc[i][j]);
    __syncthreads();
    p ^= 1;
  }

  const int T = *twp;
  const int ncol = lane & 15, moff = (lane >> 4) * 8;   // C/D layout (7.12.2)
#pragma unroll
  for (int i = 0; i < 4; ++i) {
#pragma unroll
    for (int j = 0; j < 2; ++j) {
      const int h = H0 + n0 + j * 16 + ncol;
      const float bias = bias_v[h];
      float cur[8], rate[8], mlast[8];
#pragma unroll
      for (int e = 0; e < 8; ++e) cur[e] = acc[i][j][e] + bias;
      lif_scan8(cur, T, rate, mlast);
#pragma unroll
      for (int e = 0; e < 8; ++e) {
        const long brow = B0 + m0 + i * 16 + moff + e;
        const float v = (MODE == 1) ? rate[e] : 0.5f * (rate[e] + mlast[e]);
        Out[brow * 2048 + h] = f2bf(v);
      }
    }
  }
}

// ---------------------------------------------------------------------------
// Kernel 3: out = tanh(act2 @ W3^T + b3) * action_limit   [4096, 32] f32
// grid = 32, 256 threads. Each wave: 16 rows x 32 cols (2 N-tiles).
// A staged in LDS (double-buffered); B fragments direct from bf16 W3 (L0/L2).
// ---------------------------------------------------------------------------
__global__ void __launch_bounds__(256) snn_gemm3_out(
    const unsigned short* __restrict__ S2, const unsigned short* __restrict__ W3bf,
    const float* __restrict__ b3, const float* __restrict__ alim,
    float* __restrict__ out) {
  __shared__ __align__(16) unsigned short At[2][128 * LDSROW];
  const int B0 = blockIdx.x * 128;
  const int wave = threadIdx.x >> 5, lane = threadIdx.x & 31;
  const int m0 = wave * 16;

  const v8f vzero = {0.f, 0.f, 0.f, 0.f, 0.f, 0.f, 0.f, 0.f};
  v8f acc[2];
  acc[0] = vzero; acc[1] = vzero;

  v8bf a0, a1;
  gload_bf_tile(S2, 2048, B0, 0, a0, a1);
  int p = 0;
  const int KSTEPS = 64;                       // K = 2048
  for (int kt = 0; kt < KSTEPS; ++kt) {
    stor_bf_tile(At[p], a0, a1);
    __syncthreads();
    if (kt + 1 < KSTEPS) gload_bf_tile(S2, 2048, B0, (kt + 1) * 32, a0, a1);
    v16bf af = load_fragA(At[p], m0);
#pragma unroll
    for (int j = 0; j < 2; ++j) {
      v16bf bfr = load_fragB_g(W3bf, 2048, j * 16, kt * 32);
      acc[j] = wmma_bf16(af, bfr, acc[j]);
    }
    __syncthreads();
    p ^= 1;
  }

  const int ncol = lane & 15, moff = (lane >> 4) * 8;
#pragma unroll
  for (int j = 0; j < 2; ++j) {
    const int col = j * 16 + ncol;
    const float bias = b3[col];
    const float lim = alim[col];
#pragma unroll
    for (int e = 0; e < 8; ++e) {
      const long brow = B0 + m0 + moff + e;
      out[brow * 32 + col] = tanhf(acc[j][e] + bias) * lim;
    }
  }
}

// ---------------------------------------------------------------------------
extern "C" void kernel_launch(void* const* d_in, const int* in_sizes, int n_in,
                              void* d_out, int out_size, void* d_ws, size_t ws_size,
                              hipStream_t stream) {
  const float* obs  = (const float*)d_in[0];   // [4096,256]
  const float* W1   = (const float*)d_in[1];   // [2048,256]
  const float* b1   = (const float*)d_in[2];   // [2048]
  const float* W2   = (const float*)d_in[3];   // [2048,2048]
  const float* b2   = (const float*)d_in[4];   // [2048]
  const float* W3   = (const float*)d_in[5];   // [32,2048]
  const float* b3   = (const float*)d_in[6];   // [32]
  const float* alim = (const float*)d_in[7];   // [32]
  const int*   tw   = (const int*)d_in[8];     // scalar (16)
  float* out = (float*)d_out;                  // [4096,32]

  char* ws = (char*)d_ws;
  unsigned short* W2bf  = (unsigned short*)(ws);
  unsigned short* S1    = (unsigned short*)(ws + 8388608);
  unsigned short* S2    = (unsigned short*)(ws + 25165824);
  unsigned short* obsbf = (unsigned short*)(ws + 41943040);
  unsigned short* W1bf  = (unsigned short*)(ws + 44040192);
  unsigned short* W3bf  = (unsigned short*)(ws + 45088768);

  // One-shot f32 -> bf16 conversions (hot loops stay conversion-free).
  {
    const int pW2 = (2048 * 2048) / 2, pObs = (4096 * 256) / 2,
              pW1 = (2048 * 256) / 2,  pW3 = (32 * 2048) / 2;
    cvt_f32_bf16<<<(pW2 + 255) / 256, 256, 0, stream>>>(W2, (unsigned int*)W2bf, pW2);
    cvt_f32_bf16<<<(pObs + 255) / 256, 256, 0, stream>>>(obs, (unsigned int*)obsbf, pObs);
    cvt_f32_bf16<<<(pW1 + 255) / 256, 256, 0, stream>>>(W1, (unsigned int*)W1bf, pW1);
    cvt_f32_bf16<<<(pW3 + 255) / 256, 256, 0, stream>>>(W3, (unsigned int*)W3bf, pW3);
  }

  // Layer 1: K=256 (8 k-tiles).  Layer 2: K=2048 (64 k-tiles).
  snn_gemm_lif<8, 1><<<dim3(32, 16), 256, 0, stream>>>(obsbf, 256, W1bf, 256, b1, tw, S1);
  snn_gemm_lif<64, 2><<<dim3(32, 16), 256, 0, stream>>>(S1, 2048, W2bf, 2048, b2, tw, S2);
  snn_gemm3_out<<<dim3(32), 256, 0, stream>>>(S2, W3bf, b3, alim, out);
}
